// MirrorPDHG_53377853555449
// MI455X (gfx1250) — matrix operand order repaired and checked by reading, not compiled, not run on hardware
//
#include <hip/hip_runtime.h>

typedef __attribute__((ext_vector_type(16))) __bf16 v16bf;
typedef __attribute__((ext_vector_type(4)))  __bf16 bf4;
typedef __attribute__((ext_vector_type(8)))  float  v8f;

#define N_TOK 65536
#define KNB   32
#define DIM   128
#define WAVES_PER_BLOCK 4
#define BLOCK (WAVES_PER_BLOCK * 32)

static constexpr float RHO  = 1.0f;
static constexpr float BETA = 0.5f;

struct alignas(16) WaveLds {
    float  yfpn[DIM];        // Y_from_Pnew broadcast buffer
    float  scores[KNB];      // per-k scores
    __bf16 T[KNB][DIM];      // gathered neighbor tile (bf16, 8KB)
    __bf16 xib[DIM];         // Xi in bf16 for WMMA B operand
    __bf16 pnb[KNB];         // P_new in bf16 for WMMA A operand
};

__global__ __launch_bounds__(BLOCK) void mirror_pdhg_main(
    const float* __restrict__ M, const float* __restrict__ P,
    const float* __restrict__ Y, const float* __restrict__ Lam,
    const int* __restrict__ Kset,
    float* __restrict__ Pout, float* __restrict__ LamOut,
    float* __restrict__ blockPartials)
{
    __shared__ WaveLds lds[WAVES_PER_BLOCK];
    __shared__ float wavePart[WAVES_PER_BLOCK];

    const int wave = threadIdx.x >> 5;
    const int lane = threadIdx.x & 31;
    const int n    = blockIdx.x * WAVES_PER_BLOCK + wave;
    WaveLds& w = lds[wave];

    // ---- per-lane neighbor index / transport weight (lane == k) ----
    const int   idx  = Kset[n * KNB + lane];
    const float p    = P[n * KNB + lane];
    const float logp = __logf(p);

    // each lane owns 4 consecutive columns d = c0..c0+3
    const int c0 = lane * 4;
    const float4 yv = *(const float4*)(Y   + n * DIM + c0);
    const float4 lv = *(const float4*)(Lam + n * DIM + c0);

    // ---- gather T (coalesced 512B rows, one pass) fused with f32 P-matvec ----
    float4 yfp = make_float4(0.f, 0.f, 0.f, 0.f);
    for (int k = 0; k < KNB; ++k) {
        if (k + 8 < KNB) {
            int rpf = __shfl(idx, k + 8, 32);
            __builtin_prefetch(M + (long)rpf * DIM + c0, 0, 1);  // global_prefetch_b8
        }
        const int   rk = __shfl(idx, k, 32);
        const float pk = __shfl(p,   k, 32);
        const float4 t = *(const float4*)(M + (long)rk * DIM + c0);
        yfp.x += pk * t.x; yfp.y += pk * t.y;
        yfp.z += pk * t.z; yfp.w += pk * t.w;
        bf4 tb = { (__bf16)t.x, (__bf16)t.y, (__bf16)t.z, (__bf16)t.w };
        *(bf4*)&w.T[k][c0] = tb;                                  // ds_store_b64
    }

    // ---- Xi = Lam + rho*(Y - Y_from_P), stash bf16 copy for WMMA ----
    float4 xi;
    xi.x = lv.x + RHO * (yv.x - yfp.x);
    xi.y = lv.y + RHO * (yv.y - yfp.y);
    xi.z = lv.z + RHO * (yv.z - yfp.z);
    xi.w = lv.w + RHO * (yv.w - yfp.w);
    {
        bf4 xb = { (__bf16)xi.x, (__bf16)xi.y, (__bf16)xi.z, (__bf16)xi.w };
        *(bf4*)&w.xib[c0] = xb;
    }
    __syncthreads();

    // ---- scores = T(32x128) . Xi(128) via v_wmma_f32_16x16x32_bf16 ----
    // Two 16-row halves; accumulate over 4 D-chunks of 32.
    for (int h = 0; h < 2; ++h) {
        v8f acc = {};
        const int m  = h * 16 + (lane & 15);
        const int kb = (lane < 16) ? 0 : 8;
        for (int q = 0; q < 4; ++q) {
            v16bf a, b;
            // A: rows of T, contraction = D-chunk q
            for (int t = 0; t < 8; ++t) {
                a[t]     = w.T[m][q * 32 + kb + t];
                a[8 + t] = w.T[m][q * 32 + 16 + kb + t];
            }
            // B: every column = Xi chunk (broadcast)
            const int koff = q * 32 + ((lane < 16) ? 0 : 8);
            for (int t = 0; t < 8; ++t) { b[t] = w.xib[koff + t]; b[8 + t] = w.xib[koff + 16 + t]; }
            acc = __builtin_amdgcn_wmma_f32_16x16x32_bf16(
                false, a, false, b, (short)0, acc, false, false);
        }
        // column N=0 lives on lanes 0 (M=h*16+0..7) and 16 (M=h*16+8..15)
        if ((lane & 15) == 0) {
            const int base = h * 16 + ((lane == 0) ? 0 : 8);
            for (int r = 0; r < 8; ++r) w.scores[base + r] = acc[r];
        }
    }
    __syncthreads();

    // ---- KL mirror step: P_new = softmax(log P - beta*scores) over K=32 ----
    const float logit = logp - BETA * w.scores[lane];
    float mx = logit;
    for (int o = 16; o > 0; o >>= 1) mx = fmaxf(mx, __shfl_xor(mx, o, 32));
    const float ex = __expf(logit - mx);
    float sm = ex;
    for (int o = 16; o > 0; o >>= 1) sm += __shfl_xor(sm, o, 32);
    const float pn = ex / sm;
    Pout[n * KNB + lane] = pn;
    w.pnb[lane] = (__bf16)pn;
    __syncthreads();

    // ---- Y_from_Pnew = P_new(1x32) . T(32x128) via WMMA, K=32 contraction ----
    v16bf a;
    {
        const int kb = (lane < 16) ? 0 : 8;
        for (int t = 0; t < 8; ++t) {
            a[t]     = w.pnb[kb + t];
            a[8 + t] = w.pnb[16 + kb + t];
        }
    }
    for (int q = 0; q < 8; ++q) {
        v16bf b;
        const int col  = q * 16 + (lane & 15);
        const int krow = (lane < 16) ? 0 : 8;
        for (int t = 0; t < 8; ++t) { b[t] = w.T[krow + t][col]; b[8 + t] = w.T[16 + krow + t][col]; }
        v8f c = {};
        c = __builtin_amdgcn_wmma_f32_16x16x32_bf16(
            false, a, false, b, (short)0, c, false, false);
        if (lane < 16) w.yfpn[q * 16 + lane] = c[0];  // row M=0 = result chunk
    }
    __syncthreads();

    // ---- dual ascent + energy partial ----
    const float4 yfn = *(const float4*)&w.yfpn[c0];
    float4 rn, ln;
    rn.x = yv.x - yfn.x; rn.y = yv.y - yfn.y;
    rn.z = yv.z - yfn.z; rn.w = yv.w - yfn.w;
    ln.x = lv.x + RHO * rn.x; ln.y = lv.y + RHO * rn.y;
    ln.z = lv.z + RHO * rn.z; ln.w = lv.w + RHO * rn.w;
    *(float4*)(LamOut + n * DIM + c0) = ln;

    float es = 0.5f * RHO * (rn.x * rn.x + rn.y * rn.y + rn.z * rn.z + rn.w * rn.w)
             + (ln.x * rn.x + ln.y * rn.y + ln.z * rn.z + ln.w * rn.w);
    for (int o = 16; o > 0; o >>= 1) es += __shfl_xor(es, o, 32);
    if (lane == 0) wavePart[wave] = es;
    __syncthreads();
    if (threadIdx.x == 0) {
        float b0 = 0.f;
        for (int i = 0; i < WAVES_PER_BLOCK; ++i) b0 += wavePart[i];
        blockPartials[blockIdx.x] = b0;
    }
}

// Deterministic fixed-order final energy reduction.
__global__ __launch_bounds__(256) void reduce_energy(
    const float* __restrict__ parts, int nparts, float* __restrict__ out)
{
    __shared__ float s[256];
    float acc = 0.f;
    for (int i = threadIdx.x; i < nparts; i += 256) acc += parts[i];
    s[threadIdx.x] = acc;
    __syncthreads();
    for (int st = 128; st > 0; st >>= 1) {
        if (threadIdx.x < st) s[threadIdx.x] += s[threadIdx.x + st];
        __syncthreads();
    }
    if (threadIdx.x == 0) out[0] = s[0];
}

extern "C" void kernel_launch(void* const* d_in, const int* in_sizes, int n_in,
                              void* d_out, int out_size, void* d_ws, size_t ws_size,
                              hipStream_t stream) {
    const float* M    = (const float*)d_in[0];
    const float* P    = (const float*)d_in[1];
    const float* Y    = (const float*)d_in[2];
    const float* Lam  = (const float*)d_in[3];
    const int*   Kset = (const int*)d_in[4];

    float* Pout   = (float*)d_out;                         // N*K
    float* LamOut = (float*)d_out + (size_t)N_TOK * KNB;   // N*D
    float* energy = (float*)d_out + (size_t)N_TOK * KNB + (size_t)N_TOK * DIM; // 1

    float* partials = (float*)d_ws;                        // one per block
    const int nblocks = N_TOK / WAVES_PER_BLOCK;

    mirror_pdhg_main<<<nblocks, BLOCK, 0, stream>>>(
        M, P, Y, Lam, Kset, Pout, LamOut, partials);
    reduce_energy<<<1, 256, 0, stream>>>(partials, nblocks, energy);
}
